// Fusion_27986006901195
// MI455X (gfx1250) — matrix operand order repaired and checked by reading, not compiled, not run on hardware
//
#include <hip/hip_runtime.h>
#include <hip/hip_bf16.h>
#include <math.h>

// Problem constants (match reference)
constexpr int kB = 4, kN = 1024, kC = 768, kH = 12, kD = 64, kMH = 256;
constexpr float kEPS = 1e-5f;
constexpr float kSCALE = 0.125f; // 1/sqrt(64)

typedef __attribute__((ext_vector_type(16))) _Float16 v16h;
typedef __attribute__((ext_vector_type(8)))  _Float16 v8h;
typedef __attribute__((ext_vector_type(8)))  float    v8f;

__device__ inline v8f vzero() {
  v8f z;
#pragma unroll
  for (int i = 0; i < 8; i++) z[i] = 0.f;
  return z;
}

__device__ inline v16h cat8(v8h lo, v8h hi) {
  return __builtin_shufflevector(lo, hi, 0,1,2,3,4,5,6,7,8,9,10,11,12,13,14,15);
}

// ---------------------------------------------------------------------------
// CDNA5 async global->LDS copy (ASYNCcnt tracked). One 16B chunk per lane.
// Generic LDS pointer's low 32 bits are the wave-relative LDS offset.
// ---------------------------------------------------------------------------
__device__ inline void async_load_b128(const void* gaddr, void* lds) {
  unsigned ldsoff = (unsigned)(size_t)lds;
  asm volatile("global_load_async_to_lds_b128 %0, %1, off"
               :: "v"(ldsoff), "v"(gaddr) : "memory");
}
__device__ inline void wait_async0() {
  asm volatile("s_wait_asynccnt 0" ::: "memory");
}

// A fragment (16x32 f16, MxK) from row-major f16 src (leading dim ld elements).
// ISA layout: lane holds row (lane&15); lanes 0-15: K 0-7 then 16-23,
// lanes 16-31: K 8-15 then 24-31. Two 16B loads.
__device__ inline v16h load_frag_a(const _Float16* src, int ld, int row0, int k0, int lane) {
  int r  = row0 + (lane & 15);
  int kb = k0 + ((lane >> 4) << 3);        // +8 for upper half-wave
  const _Float16* p = src + (size_t)r * ld + kb;
  v8h lo = *(const v8h*)(p);
  v8h hi = *(const v8h*)(p + 16);
  return cat8(lo, hi);
}

// B fragment (32x16 f16, KxN) from "BT" layout bt[n][k] (leading dim ldk elements).
// Lane holds column (lane&15); lanes 0-15: K 0-7 then 16-23, lanes 16-31: K 8-15 then 24-31.
__device__ inline v16h load_frag_b(const _Float16* bt, int ldk, int n0, int k0, int lane) {
  int n  = n0 + (lane & 15);
  int kb = k0 + ((lane >> 4) << 3);        // +8 for upper half-wave
  const _Float16* p = bt + (size_t)n * ldk + kb;
  v8h lo = *(const v8h*)(p);
  v8h hi = *(const v8h*)(p + 16);
  return cat8(lo, hi);
}

__device__ inline v8f wmma_f16(v16h a, v16h b, v8f c) {
  return __builtin_amdgcn_wmma_f32_16x16x32_f16(false, a, false, b, (short)0, c, false, false);
}

// ---------------------------------------------------------------------------
// Kernel 1: LayerNorm over C=768 + cast to f16. One row per 256-thread block.
// ---------------------------------------------------------------------------
__global__ __launch_bounds__(256) void ln_cast_kernel(
    const float* __restrict__ x, const float* __restrict__ w,
    const float* __restrict__ bias, _Float16* __restrict__ out) {
  int row = blockIdx.x;
  const float* xr = x + (size_t)row * kC;
  __shared__ float red[2][8];
  float vals[3];
  float s = 0.f, s2 = 0.f;
#pragma unroll
  for (int i = 0; i < 3; i++) {
    float v = xr[threadIdx.x + i * 256];
    vals[i] = v; s += v; s2 += v * v;
  }
#pragma unroll
  for (int off = 16; off > 0; off >>= 1) { s += __shfl_xor(s, off); s2 += __shfl_xor(s2, off); }
  int wid = threadIdx.x >> 5;
  if ((threadIdx.x & 31) == 0) { red[0][wid] = s; red[1][wid] = s2; }
  __syncthreads();
  if (threadIdx.x == 0) {
    float a = 0.f, b = 0.f;
#pragma unroll
    for (int i = 0; i < 8; i++) { a += red[0][i]; b += red[1][i]; }
    red[0][0] = a; red[1][0] = b;
  }
  __syncthreads();
  float mu  = red[0][0] * (1.f / kC);
  float var = red[1][0] * (1.f / kC) - mu * mu;
  float rs  = rsqrtf(var + kEPS);
#pragma unroll
  for (int i = 0; i < 3; i++) {
    int c = threadIdx.x + i * 256;
    float y = (vals[i] - mu) * rs * w[c] + bias[c];
    out[(size_t)row * kC + c] = (_Float16)y;
  }
}

// Kernel 2: transpose+cast: src[K][Nn] row-major f32 -> dst[Nn][K] f16
__global__ void tcast_kernel(const float* __restrict__ s, _Float16* __restrict__ d, int K, int Nn) {
  int i = blockIdx.x * 256 + threadIdx.x;
  if (i < K * Nn) { int k = i / Nn, n = i % Nn; d[(size_t)n * K + k] = (_Float16)s[i]; }
}

// ---------------------------------------------------------------------------
// Kernel 3: QKV GEMM: [4096,768] x [768,2304] (+bias), scatter to q(f32), k(f16)
// in [B,H,N,D] and v(f16) in transposed [B,H,D,N]. Tile 128x128, BK=32,
// double-buffered async global->LDS pipeline, 8 waves as 2x4 (each 64x32).
// ---------------------------------------------------------------------------
__global__ __launch_bounds__(256) void qkv_gemm_kernel(
    const _Float16* __restrict__ xn,   // [4096][768]
    const _Float16* __restrict__ wqT,  // [2304][768]  (BT layout)
    const float* __restrict__ qkv_b,   // [2304]
    float* __restrict__ qout,          // [B,H,N,D] f32
    _Float16* __restrict__ kout,       // [B,H,N,D] f16
    _Float16* __restrict__ vtout) {    // [B,H,D,N] f16
  __shared__ _Float16 As[2][128][40];  // row-major [m][k]
  __shared__ _Float16 Bs[2][128][40];  // BT layout [n][k]
  int tid = threadIdx.x, lane = tid & 31, wid = tid >> 5;
  int wave_m = wid >> 2, wave_n = wid & 3;
  int mBase = blockIdx.y * 128, nBase = blockIdx.x * 128;

  auto issue_tile = [&](int buf, int k0) {
    for (int i = tid; i < 128 * 4; i += 256) {
      int r = i >> 2, seg = i & 3;
      async_load_b128(xn  + (size_t)(mBase + r) * kC + k0 + seg * 8, &As[buf][r][seg * 8]);
      async_load_b128(wqT + (size_t)(nBase + r) * kC + k0 + seg * 8, &Bs[buf][r][seg * 8]);
    }
  };

  v8f acc[4][2];
#pragma unroll
  for (int mt = 0; mt < 4; mt++)
#pragma unroll
    for (int nt = 0; nt < 2; nt++) acc[mt][nt] = vzero();

  issue_tile(0, 0);
  for (int k0 = 0; k0 < kC; k0 += 32) {
    int buf = (k0 >> 5) & 1;
    wait_async0();
    __syncthreads();                       // tile `buf` visible to all waves
    if (k0 + 32 < kC) issue_tile(buf ^ 1, k0 + 32);  // prefetch next tile

    v16h a[4], bfr[2];
#pragma unroll
    for (int mt = 0; mt < 4; mt++)
      a[mt] = load_frag_a(&As[buf][0][0], 40, wave_m * 64 + mt * 16, 0, lane);
#pragma unroll
    for (int nt = 0; nt < 2; nt++)
      bfr[nt] = load_frag_b(&Bs[buf][0][0], 40, wave_n * 32 + nt * 16, 0, lane);
#pragma unroll
    for (int mt = 0; mt < 4; mt++)
#pragma unroll
      for (int nt = 0; nt < 2; nt++) acc[mt][nt] = wmma_f16(a[mt], bfr[nt], acc[mt][nt]);
    __syncthreads();                       // all frag reads done before buf reuse
  }

  // Scatter epilogue. Column tile lies fully inside one of q/k/v (768%128==0).
  int which = nBase / kC;  // 0=q,1=k,2=v
#pragma unroll
  for (int mt = 0; mt < 4; mt++) {
#pragma unroll
    for (int nt = 0; nt < 2; nt++) {
      int col  = nBase + wave_n * 32 + nt * 16 + (lane & 15);
      int rowB = mBase + wave_m * 64 + mt * 16 + ((lane >> 4) << 3);
      int cmod = col % kC;
      int h = cmod >> 6, d = cmod & 63;
      float bias = qkv_b[col];
#pragma unroll
      for (int v = 0; v < 8; v++) {
        int m = rowB + v;
        int bidx = m >> 10, n = m & 1023;
        float val = acc[mt][nt][v] + bias;
        if (which == 0) {
          qout[(((size_t)bidx * kH + h) * kN + n) * kD + d] = val;
        } else if (which == 1) {
          kout[(((size_t)bidx * kH + h) * kN + n) * kD + d] = (_Float16)val;
        } else {
          vtout[(((size_t)bidx * kH + h) * kD + d) * kN + n] = (_Float16)val;
        }
      }
    }
  }
}

// ---------------------------------------------------------------------------
// Kernel 4: flash cross-attention + proj + LN + MLP + residuals.
// Grid: (N/128, B*H). 8 waves, each owns 16 Q rows. Double-buffered async K/V.
// ---------------------------------------------------------------------------
__global__ __launch_bounds__(256) void cross_attn_kernel(
    const float*    __restrict__ qf,    // [B,H,N,D] f32
    const _Float16* __restrict__ kf,    // [B,H,N,D] f16
    const _Float16* __restrict__ vt,    // [B,H,D,N] f16 (pre-transposed)
    const _Float16* __restrict__ pwT,   // [64][64]   BT of proj_w
    const float*    __restrict__ pb,
    const float*    __restrict__ nw, const float* __restrict__ nb,
    const _Float16* __restrict__ fc1T,  // [256][64]  BT of fc1_w
    const float*    __restrict__ b1,
    const _Float16* __restrict__ fc2T,  // [64][256]  BT of fc2_w
    const float*    __restrict__ b2v,
    float*          __restrict__ outp)  // [B,N,C] slice base
{
  // SPs: Q staging first (dead after A-frags hoisted to regs), then P/O/h staging.
  __shared__ _Float16 SPs[128][72];
  __shared__ _Float16 Ks[2][64][72];    // [key][d]
  __shared__ _Float16 Vt[2][64][72];    // [d][key]

  int tid = threadIdx.x, lane = tid & 31, wid = tid >> 5;
  int bh = blockIdx.y;
  int b = bh / kH, h = bh % kH;
  int rowBase = blockIdx.x * 128;
  const float*    qbase  = qf + (size_t)bh * kN * kD;
  const _Float16* kbase  = kf + (size_t)bh * kN * kD;
  const _Float16* vtbase = vt + (size_t)bh * kD * kN;

  // Stage Q tile (f32 -> f16)
  for (int i = tid; i < 128 * 64; i += 256) {
    int r = i >> 6, d = i & 63;
    SPs[r][d] = (_Float16)qbase[(size_t)(rowBase + r) * kD + d];
  }
  __syncthreads();

  int wr = wid * 16;                   // wave's row strip within tile
  v16h aq0 = load_frag_a(&SPs[0][0], 72, wr, 0, lane);
  v16h aq1 = load_frag_a(&SPs[0][0], 72, wr, 32, lane);
  __syncthreads();                     // all Q reads done before SPs is reused

  float mstate[8], lstate[8];
  v8f o[4];
#pragma unroll
  for (int v = 0; v < 8; v++) { mstate[v] = -1e30f; lstate[v] = 0.f; }
#pragma unroll
  for (int dt = 0; dt < 4; dt++) o[dt] = vzero();

  auto issue_kv = [&](int buf, int kb0) {
    for (int i = tid; i < 64 * 8; i += 256) {
      int r = i >> 3, seg = i & 7;
      async_load_b128(kbase + (size_t)(kb0 + r) * kD + seg * 8, &Ks[buf][r][seg * 8]);
    }
    for (int i = tid; i < 64 * 8; i += 256) {
      int r = i >> 3, seg = i & 7;
      async_load_b128(vtbase + (size_t)r * kN + kb0 + seg * 8, &Vt[buf][r][seg * 8]);
    }
  };

  issue_kv(0, 0);
  for (int kb0 = 0; kb0 < kN; kb0 += 64) {
    int buf = (kb0 >> 6) & 1;
    wait_async0();
    __syncthreads();                           // tile `buf` ready everywhere
    if (kb0 + 64 < kN) issue_kv(buf ^ 1, kb0 + 64);  // prefetch next K/V block

    // S = (Q K^T) * scale   (16 rows x 64 keys per wave)
    v8f s[4];
#pragma unroll
    for (int nt = 0; nt < 4; nt++) {
      v8f z = vzero();
      v16h bk0 = load_frag_b(&Ks[buf][0][0], 72, nt * 16, 0, lane);
      v16h bk1 = load_frag_b(&Ks[buf][0][0], 72, nt * 16, 32, lane);
      z = wmma_f16(aq0, bk0, z);
      z = wmma_f16(aq1, bk1, z);
      s[nt] = z;
    }
#pragma unroll
    for (int nt = 0; nt < 4; nt++)
#pragma unroll
      for (int v = 0; v < 8; v++) s[nt][v] *= kSCALE;

    // Online softmax per row (row = wr + v + 8*half; 16 lanes hold the columns)
#pragma unroll
    for (int v = 0; v < 8; v++) {
      float mx = fmaxf(fmaxf(s[0][v], s[1][v]), fmaxf(s[2][v], s[3][v]));
#pragma unroll
      for (int off = 1; off < 16; off <<= 1) mx = fmaxf(mx, __shfl_xor(mx, off));
      float mn = fmaxf(mstate[v], mx);
      float al = __expf(mstate[v] - mn);
      float rs = 0.f;
#pragma unroll
      for (int nt = 0; nt < 4; nt++) { float p = __expf(s[nt][v] - mn); s[nt][v] = p; rs += p; }
#pragma unroll
      for (int off = 1; off < 16; off <<= 1) rs += __shfl_xor(rs, off);
      lstate[v] = lstate[v] * al + rs;
      mstate[v] = mn;
#pragma unroll
      for (int dt = 0; dt < 4; dt++) o[dt][v] *= al;
    }

    // P -> LDS (C-layout -> memory) so we can reload as A fragments (own rows only)
#pragma unroll
    for (int nt = 0; nt < 4; nt++) {
      int col = nt * 16 + (lane & 15);
#pragma unroll
      for (int v = 0; v < 8; v++) {
        int r = wr + v + ((lane >> 4) << 3);
        SPs[r][col] = (_Float16)s[nt][v];
      }
    }
    v16h ap0 = load_frag_a(&SPs[0][0], 72, wr, 0, lane);
    v16h ap1 = load_frag_a(&SPs[0][0], 72, wr, 32, lane);
#pragma unroll
    for (int dt = 0; dt < 4; dt++) {
      v16h bv0 = load_frag_b(&Vt[buf][0][0], 72, dt * 16, 0, lane);
      v16h bv1 = load_frag_b(&Vt[buf][0][0], 72, dt * 16, 32, lane);
      o[dt] = wmma_f16(ap0, bv0, o[dt]);
      o[dt] = wmma_f16(ap1, bv1, o[dt]);
    }
    // No trailing barrier: next iteration's top barrier orders buffer reuse,
    // and the compiler's pre-barrier s_wait_dscnt covers outstanding DS reads.
  }

  // O /= l
  float inv[8];
#pragma unroll
  for (int v = 0; v < 8; v++) inv[v] = (lstate[v] > 0.f) ? (1.f / lstate[v]) : 0.f;
#pragma unroll
  for (int dt = 0; dt < 4; dt++)
#pragma unroll
    for (int v = 0; v < 8; v++) o[dt][v] *= inv[v];

  // O -> LDS, reload as A; proj: qq = O @ pw + pb + q (residual)
#pragma unroll
  for (int dt = 0; dt < 4; dt++) {
    int col = dt * 16 + (lane & 15);
#pragma unroll
    for (int v = 0; v < 8; v++) {
      int r = wr + v + ((lane >> 4) << 3);
      SPs[r][col] = (_Float16)o[dt][v];
    }
  }
  v16h ao0 = load_frag_a(&SPs[0][0], 72, wr, 0, lane);
  v16h ao1 = load_frag_a(&SPs[0][0], 72, wr, 32, lane);
  v8f qq[4];
#pragma unroll
  for (int dt = 0; dt < 4; dt++) {
    v8f z = vzero();
    v16h bp0 = load_frag_b(pwT, 64, dt * 16, 0, lane);
    v16h bp1 = load_frag_b(pwT, 64, dt * 16, 32, lane);
    z = wmma_f16(ao0, bp0, z);
    z = wmma_f16(ao1, bp1, z);
    qq[dt] = z;
  }
#pragma unroll
  for (int dt = 0; dt < 4; dt++) {
    int d = dt * 16 + (lane & 15);
    float bias = pb[d];
#pragma unroll
    for (int v = 0; v < 8; v++) {
      int r = rowBase + wr + v + ((lane >> 4) << 3);
      qq[dt][v] += bias + qbase[(size_t)r * kD + d];
    }
  }

  // LayerNorm over D=64 per row
  float mu[8], rstd[8];
#pragma unroll
  for (int v = 0; v < 8; v++) {
    float sm = qq[0][v] + qq[1][v] + qq[2][v] + qq[3][v];
    float sq = qq[0][v]*qq[0][v] + qq[1][v]*qq[1][v] + qq[2][v]*qq[2][v] + qq[3][v]*qq[3][v];
#pragma unroll
    for (int off = 1; off < 16; off <<= 1) { sm += __shfl_xor(sm, off); sq += __shfl_xor(sq, off); }
    float m_ = sm * (1.f / 64.f);
    float var = sq * (1.f / 64.f) - m_ * m_;
    mu[v] = m_; rstd[v] = rsqrtf(var + kEPS);
  }
  // h = LN(qq)*nw+nb -> LDS f16
#pragma unroll
  for (int dt = 0; dt < 4; dt++) {
    int d = dt * 16 + (lane & 15);
    float w_ = nw[d], bb_ = nb[d];
#pragma unroll
    for (int v = 0; v < 8; v++) {
      float hv = (qq[dt][v] - mu[v]) * rstd[v] * w_ + bb_;
      int r = wr + v + ((lane >> 4) << 3);
      SPs[r][d] = (_Float16)hv;
    }
  }
  v16h ah0 = load_frag_a(&SPs[0][0], 72, wr, 0, lane);
  v16h ah1 = load_frag_a(&SPs[0][0], 72, wr, 32, lane);

  // MLP 64->256 (gelu) ->64, hidden processed in 4 chunks of 64
  v8f macc[4];
#pragma unroll
  for (int dt = 0; dt < 4; dt++) macc[dt] = vzero();
  for (int cc = 0; cc < 4; cc++) {
#pragma unroll
    for (int t = 0; t < 4; t++) {
      v8f z = vzero();
      v16h bf0 = load_frag_b(fc1T, 64, cc * 64 + t * 16, 0, lane);
      v16h bf1 = load_frag_b(fc1T, 64, cc * 64 + t * 16, 32, lane);
      z = wmma_f16(ah0, bf0, z);
      z = wmma_f16(ah1, bf1, z);
      int n = cc * 64 + t * 16 + (lane & 15);
      float bias = b1[n];
      int col = t * 16 + (lane & 15);
#pragma unroll
      for (int v = 0; v < 8; v++) {
        float x = z[v] + bias;
        float g = 0.5f * x * (1.f + erff(x * 0.70710678f));  // exact gelu
        int r = wr + v + ((lane >> 4) << 3);
        SPs[r][col] = (_Float16)g;
      }
    }
    v16h ag0 = load_frag_a(&SPs[0][0], 72, wr, 0, lane);
    v16h ag1 = load_frag_a(&SPs[0][0], 72, wr, 32, lane);
#pragma unroll
    for (int dt = 0; dt < 4; dt++) {
      v16h bc0 = load_frag_b(fc2T, 256, dt * 16, cc * 64, lane);
      v16h bc1 = load_frag_b(fc2T, 256, dt * 16, cc * 64 + 32, lane);
      macc[dt] = wmma_f16(ag0, bc0, macc[dt]);
      macc[dt] = wmma_f16(ag1, bc1, macc[dt]);
    }
  }

  // out = mlp + b2 + qq  -> [B,N,C] at c = h*64 + d
#pragma unroll
  for (int dt = 0; dt < 4; dt++) {
    int d = dt * 16 + (lane & 15);
    float bias = b2v[d];
#pragma unroll
    for (int v = 0; v < 8; v++) {
      int r = rowBase + wr + v + ((lane >> 4) << 3);
      outp[((size_t)b * kN + r) * kC + h * 64 + d] = macc[dt][v] + bias + qq[dt][v];
    }
  }
}

// ---------------------------------------------------------------------------
// Launcher
// ---------------------------------------------------------------------------
extern "C" void kernel_launch(void* const* d_in, const int* in_sizes, int n_in,
                              void* d_out, int out_size, void* d_ws, size_t ws_size,
                              hipStream_t stream) {
  (void)in_sizes; (void)n_in; (void)out_size; (void)ws_size;
  const float* x_vis = (const float*)d_in[0];
  const float* x_inf = (const float*)d_in[1];
  const float* n1w   = (const float*)d_in[2];
  const float* n1b   = (const float*)d_in[3];
  const float* qkvw  = (const float*)d_in[4];
  const float* qkvb  = (const float*)d_in[5];
  // c{1,2}: proj_w, proj_b, norm_w, norm_b, fc1_w, fc1_b, fc2_w, fc2_b
  const float* c1[8]; const float* c2[8];
  for (int i = 0; i < 8; i++) c1[i] = (const float*)d_in[6 + i];
  for (int i = 0; i < 8; i++) c2[i] = (const float*)d_in[14 + i];

  float* out = (float*)d_out;
  char* ws = (char*)d_ws;
  size_t off = 0;
  auto take = [&](size_t bytes) -> char* {
    char* p = ws + off;
    off = (off + bytes + 255) & ~(size_t)255;
    return p;
  };
  const size_t BNC = (size_t)kB * kN * kC;   // 3,145,728
  _Float16* xn_vis = (_Float16*)take(BNC * 2);
  _Float16* xn_inf = (_Float16*)take(BNC * 2);
  _Float16* wqT    = (_Float16*)take((size_t)3 * kC * kC * 2);  // [2304][768]
  float*    q_vis  = (float*)take(BNC * 4);
  float*    q_inf  = (float*)take(BNC * 4);
  _Float16* k_vis  = (_Float16*)take(BNC * 2);
  _Float16* vT_vis = (_Float16*)take(BNC * 2);
  _Float16* k_inf  = (_Float16*)take(BNC * 2);
  _Float16* vT_inf = (_Float16*)take(BNC * 2);
  _Float16* pwT1   = (_Float16*)take((size_t)kD * kD * 2);
  _Float16* fc1T1  = (_Float16*)take((size_t)kD * kMH * 2);
  _Float16* fc2T1  = (_Float16*)take((size_t)kMH * kD * 2);
  _Float16* pwT2   = (_Float16*)take((size_t)kD * kD * 2);
  _Float16* fc1T2  = (_Float16*)take((size_t)kD * kMH * 2);
  _Float16* fc2T2  = (_Float16*)take((size_t)kMH * kD * 2);

  // 1) LN + cast activations
  ln_cast_kernel<<<kB * kN, 256, 0, stream>>>(x_vis, n1w, n1b, xn_vis);
  ln_cast_kernel<<<kB * kN, 256, 0, stream>>>(x_inf, n1w, n1b, xn_inf);

  // 2) weight prep (all transposed to BT layout, f16)
  {
    int n = kC * 3 * kC;
    tcast_kernel<<<(n + 255) / 256, 256, 0, stream>>>(qkvw, wqT, kC, 3 * kC);
  }
  tcast_kernel<<<(kD * kD + 255) / 256, 256, 0, stream>>>(c1[0], pwT1, kD, kD);
  tcast_kernel<<<(kD * kMH + 255) / 256, 256, 0, stream>>>(c1[4], fc1T1, kD, kMH);
  tcast_kernel<<<(kMH * kD + 255) / 256, 256, 0, stream>>>(c1[6], fc2T1, kMH, kD);
  tcast_kernel<<<(kD * kD + 255) / 256, 256, 0, stream>>>(c2[0], pwT2, kD, kD);
  tcast_kernel<<<(kD * kMH + 255) / 256, 256, 0, stream>>>(c2[4], fc1T2, kD, kMH);
  tcast_kernel<<<(kMH * kD + 255) / 256, 256, 0, stream>>>(c2[6], fc2T2, kMH, kD);

  // 3) QKV GEMMs
  dim3 ggrid(3 * kC / 128, (kB * kN) / 128);  // (18, 32)
  qkv_gemm_kernel<<<ggrid, 256, 0, stream>>>(xn_vis, wqT, qkvb, q_vis, k_vis, vT_vis);
  qkv_gemm_kernel<<<ggrid, 256, 0, stream>>>(xn_inf, wqT, qkvb, q_inf, k_inf, vT_inf);

  // 4) cross attention + fused epilogues
  dim3 agrid(kN / 128, kB * kH);  // (8, 48)
  // f1 = cross(q_inf, k_vis, v_vis) with c1 weights
  cross_attn_kernel<<<agrid, 256, 0, stream>>>(
      q_inf, k_vis, vT_vis, pwT1, c1[1], c1[2], c1[3], fc1T1, c1[5], fc2T1, c1[7], out);
  // f2 = cross(q_vis, k_inf, v_inf) with c2 weights
  cross_attn_kernel<<<agrid, 256, 0, stream>>>(
      q_vis, k_inf, vT_inf, pwT2, c2[1], c2[2], c2[3], fc1T2, c2[5], fc2T2, c2[7],
      out + BNC);
}
